// MLPSpeculatorModelTied_54477365182666
// MI455X (gfx1250) — matrix operand order repaired and checked by reading, not compiled, not run on hardware
//
#include <hip/hip_runtime.h>
#include <cmath>

// ---------------------------------------------------------------------------
// MLP speculator (4 chained heads) for MI455X / gfx1250.
// - weights converted fp32->bf16 once per launch (head_w bf16 ~196MB, ~L2-resident)
// - all GEMMs via v_wmma_f32_16x16x32_bf16
// - tiles staged with global_load_async_to_lds_b128 (ASYNCcnt), 4-buffer pipeline,
//   uniform 4 async ops per wave per stage, peeled barrier-free drain.
// ---------------------------------------------------------------------------

typedef __attribute__((ext_vector_type(16))) __bf16 v16bf;
typedef __attribute__((ext_vector_type(8)))  float  v8f;

#define TB      256  // threads per block (8 waves of 32)
#define BM      128  // block tile M (whole batch)
#define BN      128  // block tile N
#define KSTEP    32  // WMMA K per step (bf16)
#define NSTAGE    4  // async pipeline buffers (power of 2)

__device__ __forceinline__ unsigned short f32_to_bf16(float f) {
  union { float f; unsigned int u; } c; c.f = f;
  unsigned int u = c.u;
  u += 0x7fffu + ((u >> 16) & 1u);   // round-to-nearest-even
  return (unsigned short)(u >> 16);
}

// 16B/lane global -> LDS direct copy, tracked by ASYNCcnt (no VGPR round trip).
__device__ __forceinline__ void async_b128(const void* g, void* lds) {
  asm volatile("global_load_async_to_lds_b128 %0, %1, off"
               :: "v"((unsigned int)(size_t)lds),
                  "v"((unsigned long long)(size_t)g)
               : "memory");
}

// ---------------------------------------------------------------------------
// fp32 -> bf16 bulk conversion (vectorized x4, grid-strided)
// ---------------------------------------------------------------------------
__global__ void cvt_f32_bf16_kernel(const float4* __restrict__ src,
                                    ushort4* __restrict__ dst, size_t n4) {
  size_t i  = (size_t)blockIdx.x * blockDim.x + threadIdx.x;
  size_t st = (size_t)gridDim.x * blockDim.x;
  for (; i < n4; i += st) {
    float4 f = src[i];
    ushort4 o;
    o.x = f32_to_bf16(f.x); o.y = f32_to_bf16(f.y);
    o.z = f32_to_bf16(f.z); o.w = f32_to_bf16(f.w);
    dst[i] = o;
  }
}

__global__ void cvt_idx_kernel(const long long* __restrict__ src,
                               int* __restrict__ dst, int n) {
  int i = threadIdx.x;
  if (i < n) dst[i] = (int)src[i];
}

// ---------------------------------------------------------------------------
// bf16 NT GEMM: out[128 x N] = A[128 x K] * Bw[N x K]^T  (both K-contiguous)
// mode 0: out = acc
// mode 1: out = acc * state_weight + emb_w[ind[m]][n] * emb_weight
//
// LDS holds fragments pre-swizzled to the ISA §7.12.2 lane layouts:
//  A (16x32 bf16): lane L -> m = L&15, kg = L>>4;
//                  elem e -> K = (e&7) + kg*8 + (e>>3)*16
//  B (32x16 bf16): lane L -> n = L&15, kg = L>>4;
//                  elem e -> K = (e&7) + kg*8 + (e>>3)*16
// so each lane's 16-half fragment is one contiguous 32B LDS read.
// ---------------------------------------------------------------------------
__global__ __launch_bounds__(TB) void gemm_bf16_kernel(
    const unsigned short* __restrict__ A,   // [128 x K] bf16
    const unsigned short* __restrict__ Bw,  // [N x K]   bf16
    float* __restrict__ out,                // [128 x N] fp32
    int N, int K, int mode, float state_weight,
    const float* __restrict__ emb_w, const int* __restrict__ ind,
    float emb_weight)
{
  __shared__ __align__(32) unsigned short lA[NSTAGE][8][32][16]; // 32 KB
  __shared__ __align__(32) unsigned short lB[NSTAGE][8][32][16]; // 32 KB

  const int tid  = threadIdx.x;
  const int lane = tid & 31;
  const int wave = tid >> 5;
  const int n0   = blockIdx.x * BN;
  const int nk   = K / KSTEP;

  // --- fill mapping: every wave issues exactly 4 async b128 per stage ---
  // A: wave w fills M-tile w (2 instructions)
  const int a_row = wave * 16 + (lane & 15);
  const int a_kg  = lane >> 4;
  const unsigned short* aSrc = A + (size_t)a_row * K + a_kg * 8;
  // B: wave w fills N-tile w (2 instructions)
  const int b_row = n0 + wave * 16 + (lane & 15);
  const int b_kg  = lane >> 4;
  const unsigned short* bSrc = Bw + (size_t)b_row * K + b_kg * 8;

  auto fill = [&](int ks, int buf) {
    const int k0 = ks * KSTEP;
    async_b128(aSrc + k0,      &lA[buf][wave][lane][0]);   // K: kg*8 + 0..7
    async_b128(aSrc + k0 + 16, &lA[buf][wave][lane][8]);   // K: 16 + kg*8 + 0..7
    async_b128(bSrc + k0,      &lB[buf][wave][lane][0]);   // K: kg*8 + 0..7
    async_b128(bSrc + k0 + 16, &lB[buf][wave][lane][8]);   // K: 16 + kg*8 + 0..7
  };

  // --- compute mapping: 8 waves -> 4 (M) x 2 (N); each wave 2x4 tiles ---
  const int wm = wave & 3;   // M-tiles {2*wm, 2*wm+1}
  const int wn = wave >> 2;  // N-tiles {4*wn .. 4*wn+3}

  v8f acc[2][4] = {};

  auto compute = [&](int buf) {
    v16bf fa0 = *(const v16bf*)&lA[buf][wm * 2 + 0][lane][0];
    v16bf fa1 = *(const v16bf*)&lA[buf][wm * 2 + 1][lane][0];
#pragma unroll
    for (int ni = 0; ni < 4; ++ni) {
      v16bf fb = *(const v16bf*)&lB[buf][wn * 4 + ni][lane][0];
      acc[0][ni] = __builtin_amdgcn_wmma_f32_16x16x32_bf16(false, fa0, false, fb,
                                                           (short)0, acc[0][ni], false, false);
      acc[1][ni] = __builtin_amdgcn_wmma_f32_16x16x32_bf16(false, fa1, false, fb,
                                                           (short)0, acc[1][ni], false, false);
    }
  };

  // prologue: keep NSTAGE-1 stages in flight
  const int pro = (nk < NSTAGE - 1) ? nk : (NSTAGE - 1);
  for (int s = 0; s < pro; ++s) fill(s, s);

  // steady state: every step issues a fill; constant-immediate wait is exact.
  // Outstanding before wait: stages {ks..ks+2} = 12 ops; <=8 left => stage ks done.
  const int body = (nk > NSTAGE - 1) ? (nk - (NSTAGE - 1)) : 0;
  for (int ks = 0; ks < body; ++ks) {
    asm volatile("s_wait_asynccnt 0x8" ::: "memory");
    __syncthreads();                 // stage ks visible; all waves done with ks-1
    fill(ks + NSTAGE - 1, (ks + NSTAGE - 1) & (NSTAGE - 1)); // reuses stage ks-1 buf
    if (ks + 8 < nk)
      __builtin_prefetch(bSrc + (ks + 8) * KSTEP, 0, 1);     // warm L2 for streamed B
    compute(ks & (NSTAGE - 1));
  }

  // drain: remaining stages are resident; no more LDS writes -> no barriers needed
  asm volatile("s_wait_asynccnt 0x0" ::: "memory");
  __syncthreads();
  for (int ks = body; ks < nk; ++ks) compute(ks & (NSTAGE - 1));

  // --- epilogue: C/D layout: lane L -> n = L&15; VGPR r -> m = r + 8*(L>>4)
  const int colIT  = lane & 15;
  const int rowOff = (lane >> 4) * 8;
#pragma unroll
  for (int mi = 0; mi < 2; ++mi) {
#pragma unroll
    for (int ni = 0; ni < 4; ++ni) {
      const int mt = wm * 2 + mi, nt = wn * 4 + ni;
#pragma unroll
      for (int r = 0; r < 8; ++r) {
        const int m = mt * 16 + rowOff + r;
        const int n = n0 + nt * 16 + colIT;
        float v = acc[mi][ni][r];
        if (mode == 1)
          v = v * state_weight + emb_w[(size_t)ind[m] * (size_t)N + n] * emb_weight;
        out[(size_t)m * N + n] = v;
      }
    }
  }
}

// ---------------------------------------------------------------------------
// RMS-LN + exact GeLU, writes bf16 state for next GEMM. One block per row.
// ---------------------------------------------------------------------------
__global__ __launch_bounds__(TB) void ln_gelu_kernel(
    const float* __restrict__ x_in, const float* __restrict__ w,
    const float* __restrict__ b, unsigned short* __restrict__ out_bf, int n)
{
  const int row = blockIdx.x;
  const float* x = x_in + (size_t)row * n;
  __shared__ float red[TB];
  float s = 0.f;
  for (int j = threadIdx.x; j < n; j += TB) { float v = x[j]; s += v * v; }
  red[threadIdx.x] = s;
  __syncthreads();
  for (int off = TB / 2; off > 0; off >>= 1) {
    if (threadIdx.x < off) red[threadIdx.x] += red[threadIdx.x + off];
    __syncthreads();
  }
  const float r = rsqrtf(red[0] / (float)n + 1e-6f);
  for (int j = threadIdx.x; j < n; j += TB) {
    float y = x[j] * r * w[j] + b[j];
    float g = 0.5f * y * (1.0f + erff(y * 0.70710678118654752f));
    out_bf[(size_t)row * n + j] = f32_to_bf16(g);
  }
}

// ---------------------------------------------------------------------------
// softmax (== exp(log_softmax)) + argmax (first-max index). One block per row.
// ---------------------------------------------------------------------------
__global__ __launch_bounds__(TB) void softmax_argmax_kernel(
    const float* __restrict__ logits, float* __restrict__ probs,
    int* __restrict__ ind, int iter, int n)
{
  const int row = blockIdx.x;
  const float* x = logits + (size_t)row * n;
  float* o = probs + ((size_t)row * 4 + iter) * (size_t)n;
  __shared__ float sv[TB];
  __shared__ int   si[TB];

  float bm = -3.4e38f; int bi = 0x7fffffff;
  for (int j = threadIdx.x; j < n; j += TB) {
    float v = x[j];
    if (v > bm) { bm = v; bi = j; }
  }
  sv[threadIdx.x] = bm; si[threadIdx.x] = bi;
  __syncthreads();
  for (int off = TB / 2; off > 0; off >>= 1) {
    if (threadIdx.x < off) {
      float v2 = sv[threadIdx.x + off]; int i2 = si[threadIdx.x + off];
      if (v2 > sv[threadIdx.x] ||
          (v2 == sv[threadIdx.x] && i2 < si[threadIdx.x])) {
        sv[threadIdx.x] = v2; si[threadIdx.x] = i2;
      }
    }
    __syncthreads();
  }
  const float mx = sv[0];
  const int amax = si[0];
  __syncthreads();

  float s = 0.f;
  for (int j = threadIdx.x; j < n; j += TB) s += expf(x[j] - mx);
  sv[threadIdx.x] = s;
  __syncthreads();
  for (int off = TB / 2; off > 0; off >>= 1) {
    if (threadIdx.x < off) sv[threadIdx.x] += sv[threadIdx.x + off];
    __syncthreads();
  }
  const float inv = 1.0f / sv[0];
  for (int j = threadIdx.x; j < n; j += TB) o[j] = expf(x[j] - mx) * inv;
  if (threadIdx.x == 0) ind[row] = amax;
}

// ---------------------------------------------------------------------------
// host orchestration
// ---------------------------------------------------------------------------
extern "C" void kernel_launch(void* const* d_in, const int* in_sizes, int n_in,
                              void* d_out, int out_size, void* d_ws, size_t ws_size,
                              hipStream_t stream) {
  (void)in_sizes; (void)n_in; (void)out_size; (void)ws_size;

  const int B_ = 128, HIDDEN = 4096, INNER = 3072, VOCAB = 32000;

  const float*     hidden    = (const float*)d_in[0];
  const long long* input_ids = (const long long*)d_in[1];
  const float*     emb_w     = (const float*)d_in[2];
  const float*     proj0_w   = (const float*)d_in[3];
  const float*     proj1_w   = (const float*)d_in[4];
  const float*     head_w    = (const float*)d_in[5];
  const float*     ln_w      = (const float*)d_in[6];
  const float*     ln_b      = (const float*)d_in[7];
  float*           out       = (float*)d_out;

  // workspace carve-up (256B aligned); total ~260 MB
  auto aln = [](size_t x) { return (x + 255) & ~(size_t)255; };
  unsigned char* p = (unsigned char*)d_ws;
  unsigned short* head_bf  = (unsigned short*)p; p += aln((size_t)VOCAB * INNER * 2);
  unsigned short* proj0_bf = (unsigned short*)p; p += aln((size_t)INNER * HIDDEN * 2);
  unsigned short* proj1_bf = (unsigned short*)p; p += aln((size_t)INNER * INNER * 2);
  unsigned short* stateA   = (unsigned short*)p; p += aln((size_t)B_ * HIDDEN * 2);
  float*          tmp      = (float*)p;          p += aln((size_t)B_ * INNER * 4);
  float*          logits   = (float*)p;          p += aln((size_t)B_ * VOCAB * 4);
  int*            ind      = (int*)p;            p += aln((size_t)B_ * 4);

  // scalar constants from the reference
  const double sw_d = pow(0.5, 0.5 / 4.0);
  const float  state_weight = (float)sw_d;
  const float  emb_weight   = (float)(sqrt(1.0 - sw_d * sw_d) * sqrt((double)INNER / 2.0));

  // 1) convert weights & initial state to bf16; copy indices
  cvt_f32_bf16_kernel<<<8192, TB, 0, stream>>>(
      (const float4*)head_w, (ushort4*)head_bf, (size_t)VOCAB * INNER / 4);
  cvt_f32_bf16_kernel<<<2048, TB, 0, stream>>>(
      (const float4*)proj0_w, (ushort4*)proj0_bf, (size_t)INNER * HIDDEN / 4);
  cvt_f32_bf16_kernel<<<2048, TB, 0, stream>>>(
      (const float4*)proj1_w, (ushort4*)proj1_bf, (size_t)INNER * INNER / 4);
  cvt_f32_bf16_kernel<<<128, TB, 0, stream>>>(
      (const float4*)hidden, (ushort4*)stateA, (size_t)B_ * HIDDEN / 4);
  cvt_idx_kernel<<<1, 128, 0, stream>>>(input_ids, ind, B_);

  // 2) four chained speculator steps
  for (int i = 0; i < 4; ++i) {
    const unsigned short* W = (i == 0) ? proj0_bf : proj1_bf;
    const int K = (i == 0) ? HIDDEN : INNER;

    // state = state @ W^T * sw + emb_w[ind] * ew   -> tmp (fp32)
    gemm_bf16_kernel<<<INNER / BN, TB, 0, stream>>>(
        stateA, W, tmp, INNER, K, /*mode=*/1, state_weight, emb_w, ind, emb_weight);

    // state = gelu(rmsnorm(tmp))  -> stateA (bf16)
    ln_gelu_kernel<<<B_, TB, 0, stream>>>(tmp, ln_w, ln_b, stateA, INNER);

    // logits = state @ head_w^T   -> logits (fp32)
    gemm_bf16_kernel<<<VOCAB / BN, TB, 0, stream>>>(
        stateA, head_bf, logits, VOCAB, INNER, /*mode=*/0, 0.f, nullptr, nullptr, 0.f);

    // probs -> out[:, i, :], ind = argmax
    softmax_argmax_kernel<<<B_, TB, 0, stream>>>(logits, out, ind, i, VOCAB);
  }
}
